// Hydra1DNeighborhoodAttention_12704513262361
// MI455X (gfx1250) — matrix operand order repaired and checked by reading, not compiled, not run on hardware
//
#include <hip/hip_runtime.h>
#include <hip/hip_bf16.h>

// ---- problem constants (from reference) ----
#define BB 4
#define TT 8192
#define CC 1024
#define NH 16
#define HS 64
#define MROWS (BB * TT)  // 32768

typedef __attribute__((ext_vector_type(16))) __bf16 v16bf;
typedef __attribute__((ext_vector_type(8)))  __bf16 v8bf;
typedef __attribute__((ext_vector_type(8)))  float  v8f;

// ---------------- f32 -> bf16 convert ----------------
__global__ void cvt_f32_bf16(const float* __restrict__ in, __bf16* __restrict__ out, long n) {
  long i = (long)blockIdx.x * blockDim.x + threadIdx.x;
  long stride = (long)gridDim.x * blockDim.x;
  for (; i < n; i += stride) out[i] = (__bf16)in[i];
}

// ---------------- WMMA bf16 GEMM:  C[m,n] = sum_k A[m,k]*W[n,k] + bias[n] ----------------
// A: M x K bf16 row-major; W: N x K bf16 row-major (W^T multiply is K-contiguous for both).
// One wave computes a 32x64 tile: 2 A fragments x 4 B fragments = 8 WMMA accumulators.
// Per 32-deep k-step: 12 b128 loads feed 8 v_wmma_f32_16x16x32_bf16.
__device__ inline v16bf load_frag_pair(const __bf16* p0, const __bf16* p1) {
  v8bf lo = *(const v8bf*)p0;
  v8bf hi = *(const v8bf*)(p1);
  return __builtin_shufflevector(lo, hi, 0,1,2,3,4,5,6,7,8,9,10,11,12,13,14,15);
}

__global__ __launch_bounds__(256) void gemm_bf16_wmma(
    const __bf16* __restrict__ A, const __bf16* __restrict__ W,
    const float* __restrict__ bias, float* __restrict__ Cout,
    int M, int N, int K) {
  const int lane = threadIdx.x & 31;
  const int wave = threadIdx.x >> 5;
  const int tiles_n = N / 64;
  const int tile = blockIdx.x * 8 + wave;
  const int tm = tile / tiles_n;
  const int tn = tile - tm * tiles_n;
  if (tm * 32 >= M) return;
  const int m0 = tm * 32;
  const int n0 = tn * 64;

  v8f acc00 = {}, acc01 = {}, acc02 = {}, acc03 = {};
  v8f acc10 = {}, acc11 = {}, acc12 = {}, acc13 = {};

  // A fragment addressing per ISA 16-bit A 16x32 layout:
  // lanes 0-15: row m0+lane, K {0..7, 16..23}; lanes 16-31: row m0+lane-16, K {8..15, 24..31}
  const int ka = (lane & 16) ? 8 : 0;
  const __bf16* aptr0 = A + (size_t)(m0 + (lane & 15)) * K + ka;
  const __bf16* aptr1 = aptr0 + (size_t)16 * K;

  // B fragment addressing: lane holds column n0+lane%16; lanes 0-15 K {0..7, 16..23}, lanes 16-31 K {8..15, 24..31}
  const int nc = lane & 15;
  const int kb = (lane & 16) ? 8 : 0;
  const __bf16* wp0 = W + (size_t)(n0 + nc) * K + kb;
  const __bf16* wp1 = wp0 + (size_t)16 * K;
  const __bf16* wp2 = wp0 + (size_t)32 * K;
  const __bf16* wp3 = wp0 + (size_t)48 * K;

  for (int kk = 0; kk < K; kk += 32) {
    __builtin_prefetch(aptr0 + kk + 256, 0, 3);  // near-scope global_prefetch_b8
    v16bf af0 = load_frag_pair(aptr0 + kk, aptr0 + kk + 16);
    v16bf af1 = load_frag_pair(aptr1 + kk, aptr1 + kk + 16);
    v16bf b0  = load_frag_pair(wp0 + kk, wp0 + kk + 16);
    v16bf b1  = load_frag_pair(wp1 + kk, wp1 + kk + 16);
    v16bf b2  = load_frag_pair(wp2 + kk, wp2 + kk + 16);
    v16bf b3  = load_frag_pair(wp3 + kk, wp3 + kk + 16);
    acc00 = __builtin_amdgcn_wmma_f32_16x16x32_bf16(false, af0, false, b0, (short)0, acc00, false, false);
    acc01 = __builtin_amdgcn_wmma_f32_16x16x32_bf16(false, af0, false, b1, (short)0, acc01, false, false);
    acc02 = __builtin_amdgcn_wmma_f32_16x16x32_bf16(false, af0, false, b2, (short)0, acc02, false, false);
    acc03 = __builtin_amdgcn_wmma_f32_16x16x32_bf16(false, af0, false, b3, (short)0, acc03, false, false);
    acc10 = __builtin_amdgcn_wmma_f32_16x16x32_bf16(false, af1, false, b0, (short)0, acc10, false, false);
    acc11 = __builtin_amdgcn_wmma_f32_16x16x32_bf16(false, af1, false, b1, (short)0, acc11, false, false);
    acc12 = __builtin_amdgcn_wmma_f32_16x16x32_bf16(false, af1, false, b2, (short)0, acc12, false, false);
    acc13 = __builtin_amdgcn_wmma_f32_16x16x32_bf16(false, af1, false, b3, (short)0, acc13, false, false);
  }

  // C/D layout: VGPR e -> row +e +8*(lane>=16); col n0 + lane%16 (+16 per n sub-tile)
  const int rb = m0 + ((lane & 16) ? 8 : 0);
  const float bia0 = bias[n0 + nc];
  const float bia1 = bias[n0 + 16 + nc];
  const float bia2 = bias[n0 + 32 + nc];
  const float bia3 = bias[n0 + 48 + nc];
#pragma unroll
  for (int e = 0; e < 8; ++e) {
    float* crow0 = Cout + (size_t)(rb + e) * N + n0 + nc;
    float* crow1 = crow0 + (size_t)16 * N;
    crow0[0]  = acc00[e] + bia0;
    crow0[16] = acc01[e] + bia1;
    crow0[32] = acc02[e] + bia2;
    crow0[48] = acc03[e] + bia3;
    crow1[0]  = acc10[e] + bia0;
    crow1[16] = acc11[e] + bia1;
    crow1[32] = acc12[e] + bia2;
    crow1[48] = acc13[e] + bia3;
  }
}

// ---------------- scatter pre-RoPE k/v into present layout [b][h][t][d] ----------------
__global__ void scatter_kv(const float* __restrict__ src, float* __restrict__ dst, long n) {
  long i = (long)blockIdx.x * blockDim.x + threadIdx.x;
  long stride = (long)gridDim.x * blockDim.x;
  for (; i < n; i += stride) {
    int dd = (int)(i % HS);
    long tmp = i / HS;
    int t = (int)(tmp % TT);
    tmp /= TT;
    int h = (int)(tmp % NH);
    int b = (int)(tmp / NH);
    dst[i] = src[((size_t)b * TT + t) * CC + h * HS + dd];
  }
}

// ---------------- RoPE (first 32 dims of each head), in place ----------------
__global__ void rope_kernel(float* __restrict__ buf) {
  long idx = (long)blockIdx.x * blockDim.x + threadIdx.x;
  const long total = (long)MROWS * NH * 16;
  if (idx >= total) return;
  int i = (int)(idx % 16);
  int h = (int)((idx / 16) % NH);
  long row = idx / (16 * NH);
  int t = (int)(row % TT);
  float inv = __powf(10000.f, -(float)(2 * i) / 32.f);
  float fr = (float)t * inv;
  float c = __cosf(fr), s = __sinf(fr);
  float* pp = buf + (size_t)row * CC + h * HS + 2 * i;
  float x1 = pp[0], x2 = pp[1];
  pp[0] = x1 * c - x2 * s;
  pp[1] = x2 * c + x1 * s;
}

// ---------------- neighborhood attention: one wave per (b,h,t), lane = 2 dims ----------------
__global__ __launch_bounds__(256) void neigh_attn(
    const float* __restrict__ q, const float* __restrict__ k,
    const float* __restrict__ v, float* __restrict__ out) {
  const int lane = threadIdx.x & 31;
  long wid = (long)blockIdx.x * 8 + (threadIdx.x >> 5);  // [0, B*NH*T)
  int t = (int)(wid % TT);
  int h = (int)((wid / TT) % NH);
  int b = (int)(wid / ((long)TT * NH));
  const int dil_tab[4] = {1, 4, 8, 16};
  int d = dil_tab[h >> 2];
  int r = t % d, p = t / d;
  int Lr = (TT - r + d - 1) / d;
  int s = p - 3;
  if (s < 0) s = 0;
  if (s > Lr - 7) s = Lr - 7;

  const size_t base = ((size_t)b * TT) * CC + (size_t)h * HS + lane * 2;
  const float2 qv = *(const float2*)(q + base + (size_t)t * CC);

  float sc[7];
  int idxs[7];
#pragma unroll
  for (int j = 0; j < 7; ++j) {
    int nt = r + (s + j) * d;
    idxs[j] = nt;
    float2 kv = *(const float2*)(k + base + (size_t)nt * CC);
    float part = qv.x * kv.x + qv.y * kv.y;
#pragma unroll
    for (int off = 16; off; off >>= 1) part += __shfl_xor(part, off, 32);
    sc[j] = part;  // all lanes hold the full dot product
  }
  float m = sc[0];
#pragma unroll
  for (int j = 1; j < 7; ++j) m = fmaxf(m, sc[j]);
  float den = 0.f;
#pragma unroll
  for (int j = 0; j < 7; ++j) { sc[j] = __expf(sc[j] - m); den += sc[j]; }
  float inv = 1.f / den;
  float2 accv = {0.f, 0.f};
#pragma unroll
  for (int j = 0; j < 7; ++j) {
    float2 vv = *(const float2*)(v + base + (size_t)idxs[j] * CC);
    float w = sc[j] * inv;
    accv.x += w * vv.x;
    accv.y += w * vv.y;
  }
  *(float2*)(out + base + (size_t)t * CC) = accv;
}

// ---------------- launch ----------------
extern "C" void kernel_launch(void* const* d_in, const int* in_sizes, int n_in,
                              void* d_out, int out_size, void* d_ws, size_t ws_size,
                              hipStream_t stream) {
  const float* x  = (const float*)d_in[0];
  const float* wq = (const float*)d_in[1];
  const float* bq = (const float*)d_in[2];
  const float* wk = (const float*)d_in[3];
  const float* bk = (const float*)d_in[4];
  const float* wv = (const float*)d_in[5];
  const float* bv = (const float*)d_in[6];
  const float* wp = (const float*)d_in[7];
  const float* bp = (const float*)d_in[8];
  float* out = (float*)d_out;

  char* ws = (char*)d_ws;
  size_t off = 0;
  auto alloc = [&](size_t bytes) -> void* {
    void* p = ws + off;
    off = (off + bytes + 255) & ~(size_t)255;
    return p;
  };
  __bf16* x_bf   = (__bf16*)alloc((size_t)MROWS * CC * 2);
  __bf16* wq_bf  = (__bf16*)alloc((size_t)CC * CC * 2);
  __bf16* wk_bf  = (__bf16*)alloc((size_t)CC * CC * 2);
  __bf16* wv_bf  = (__bf16*)alloc((size_t)CC * CC * 2);
  __bf16* wp_bf  = (__bf16*)alloc((size_t)CC * CC * 2);
  float*  qf     = (float*)alloc((size_t)MROWS * CC * 4);
  float*  kf     = (float*)alloc((size_t)MROWS * CC * 4);
  float*  vf     = (float*)alloc((size_t)MROWS * CC * 4);
  float*  yat    = (float*)alloc((size_t)MROWS * CC * 4);
  __bf16* yat_bf = (__bf16*)alloc((size_t)MROWS * CC * 2);

  const long nx = (long)MROWS * CC;
  const long nw = (long)CC * CC;
  cvt_f32_bf16<<<2048, 256, 0, stream>>>(x, x_bf, nx);
  cvt_f32_bf16<<<1024, 256, 0, stream>>>(wq, wq_bf, nw);
  cvt_f32_bf16<<<1024, 256, 0, stream>>>(wk, wk_bf, nw);
  cvt_f32_bf16<<<1024, 256, 0, stream>>>(wv, wv_bf, nw);
  cvt_f32_bf16<<<1024, 256, 0, stream>>>(wp, wp_bf, nw);

  const int tiles = (MROWS / 32) * (CC / 64);  // 16384 wave-tiles (32x64 each)
  const int gblocks = tiles / 8;               // 8 waves per 256-thread block
  gemm_bf16_wmma<<<gblocks, 256, 0, stream>>>(x_bf, wq_bf, bq, qf, MROWS, CC, CC);
  gemm_bf16_wmma<<<gblocks, 256, 0, stream>>>(x_bf, wk_bf, bk, kf, MROWS, CC, CC);
  gemm_bf16_wmma<<<gblocks, 256, 0, stream>>>(x_bf, wv_bf, bv, vf, MROWS, CC, CC);

  // present = stack([k, v]) BEFORE rope
  const long npr = (long)BB * NH * TT * HS;
  float* pres = out + (size_t)BB * TT * CC;
  scatter_kv<<<4096, 256, 0, stream>>>(kf, pres, npr);
  scatter_kv<<<4096, 256, 0, stream>>>(vf, pres + npr, npr);

  const long nrope = (long)MROWS * NH * 16;
  const int rblocks = (int)((nrope + 255) / 256);
  rope_kernel<<<rblocks, 256, 0, stream>>>(qf);
  rope_kernel<<<rblocks, 256, 0, stream>>>(kf);
  rope_kernel<<<rblocks, 256, 0, stream>>>(vf);

  const long waves = (long)BB * NH * TT;  // 524288
  neigh_attn<<<(int)(waves / 8), 256, 0, stream>>>(qf, kf, vf, yat);

  cvt_f32_bf16<<<2048, 256, 0, stream>>>(yat, yat_bf, nx);
  gemm_bf16_wmma<<<gblocks, 256, 0, stream>>>(yat_bf, wp_bf, bp, out, MROWS, CC, CC);
}
